// SequenceMixingSpectralModel_24421184045499
// MI455X (gfx1250) — compile-verified, matrix operand order checked by reading, not run on hardware
//
#include <hip/hip_runtime.h>
#include <hip/hip_bf16.h>

// ---------------------------------------------------------------------------
// SequenceMixingSpectralModel, algebraically collapsed:
//   irfft(rfft(h)*f[d]) == f[d]*h  (filter constant over frequency axis)
//   => h = embed[x] * (1+f);  LN;  logits = h @ W_out^T + b
//   => result depends only on token id (V=256): precompute 256x256 logits
//      table with split-f16 WMMA, then gather rows into the 64 MiB output.
// ---------------------------------------------------------------------------

typedef __attribute__((ext_vector_type(16))) _Float16 v16h;
typedef __attribute__((ext_vector_type(8)))  _Float16 v8h;
typedef __attribute__((ext_vector_type(8)))  float    v8f;
typedef __attribute__((ext_vector_type(4)))  float    f4;

#define D_DIM 256
#define V_DIM 256
#define NROWS 65536      /* B*L = 8*8192 */
#define LN_EPS 1e-5f

// ---------------------------------------------------------------------------
// Kernel 1: prep. Blocks 0..255: per-token scaled-embed + LayerNorm -> split
// f16 (hi/lo). Blocks 256..511: split W_out rows into f16 hi/lo.
// ---------------------------------------------------------------------------
__global__ __launch_bounds__(256) void prep_kernel(
    const float* __restrict__ embed, const float* __restrict__ freq,
    const float* __restrict__ ln_w, const float* __restrict__ ln_b,
    const float* __restrict__ W_out,
    _Float16* __restrict__ hn_hi, _Float16* __restrict__ hn_lo,
    _Float16* __restrict__ w_hi, _Float16* __restrict__ w_lo)
{
    __shared__ float ssum[256];
    __shared__ float ssq[256];
    const int d = threadIdx.x;
    const int b = blockIdx.x;

    if (b < 256) {
        const int t = b;
        float h = embed[t * D_DIM + d] * (1.0f + freq[d]);   // spectral block collapsed
        ssum[d] = h;
        ssq[d]  = h * h;
        __syncthreads();
        #pragma unroll
        for (int s = 128; s > 0; s >>= 1) {
            if (d < s) { ssum[d] += ssum[d + s]; ssq[d] += ssq[d + s]; }
            __syncthreads();
        }
        const float mu  = ssum[0] * (1.0f / 256.0f);
        const float var = ssq[0] * (1.0f / 256.0f) - mu * mu;
        const float inv = rsqrtf(var + LN_EPS);
        const float hn  = (h - mu) * inv * ln_w[d] + ln_b[d];
        const _Float16 hi = (_Float16)hn;
        const _Float16 lo = (_Float16)(hn - (float)hi);
        hn_hi[t * D_DIM + d] = hi;
        hn_lo[t * D_DIM + d] = lo;
    } else {
        const int v = b - 256;
        const float w = W_out[v * D_DIM + d];
        const _Float16 hi = (_Float16)w;
        const _Float16 lo = (_Float16)(w - (float)hi);
        w_hi[v * D_DIM + d] = hi;
        w_lo[v * D_DIM + d] = lo;
    }
}

// ---------------------------------------------------------------------------
// Fragment load per ISA 7.12.2 (16-bit A 16x32 / B-as-NxK layout):
// lane holds row (lane&15); halves 0..7 at K=sel, halves 8..15 at K=sel+16,
// sel = (lane>=16)*8. Two contiguous 16B loads.
// ---------------------------------------------------------------------------
__device__ inline v16h load_frag(const _Float16* __restrict__ p)
{
    v8h lo = *(const v8h*)(p);
    v8h hi = *(const v8h*)(p + 16);
    v16h r;
    #pragma unroll
    for (int i = 0; i < 8; ++i) { r[i] = lo[i]; r[i + 8] = hi[i]; }
    return r;
}

// ---------------------------------------------------------------------------
// Kernel 2: 256x256x256 GEMM -> logits table, split-f16 WMMA (3 products,
// lo*lo dropped => ~22-bit mantissa, fp32 accumulate). One wave per 16x16
// tile, 8 waves per block, 32 blocks.
// ---------------------------------------------------------------------------
__global__ __launch_bounds__(256) void table_gemm_kernel(
    const _Float16* __restrict__ hn_hi, const _Float16* __restrict__ hn_lo,
    const _Float16* __restrict__ w_hi, const _Float16* __restrict__ w_lo,
    const float* __restrict__ b_out, float* __restrict__ table)
{
    const int wave = threadIdx.x >> 5;
    const int lane = threadIdx.x & 31;
    const int tile = blockIdx.x * 8 + wave;        // 256 tiles total
    const int t0 = (tile >> 4) << 4;               // token tile origin (M)
    const int v0 = (tile & 15) << 4;               // vocab tile origin (N)

    const int row = lane & 15;
    const int sel = (lane >> 4) << 3;              // 0 or 8

    const _Float16* arow_hi = hn_hi + (size_t)(t0 + row) * D_DIM;
    const _Float16* arow_lo = hn_lo + (size_t)(t0 + row) * D_DIM;
    const _Float16* brow_hi = w_hi  + (size_t)(v0 + row) * D_DIM;   // W_out is N x K
    const _Float16* brow_lo = w_lo  + (size_t)(v0 + row) * D_DIM;

    v8f c = {};
    #pragma unroll
    for (int k0 = 0; k0 < D_DIM; k0 += 32) {
        v16h ah = load_frag(arow_hi + k0 + sel);
        v16h al = load_frag(arow_lo + k0 + sel);
        v16h bh = load_frag(brow_hi + k0 + sel);
        v16h bl = load_frag(brow_lo + k0 + sel);
        c = __builtin_amdgcn_wmma_f32_16x16x32_f16(false, ah, false, bh,
                                                   (short)0, c, false, false);
        c = __builtin_amdgcn_wmma_f32_16x16x32_f16(false, ah, false, bl,
                                                   (short)0, c, false, false);
        c = __builtin_amdgcn_wmma_f32_16x16x32_f16(false, al, false, bh,
                                                   (short)0, c, false, false);
    }

    // C/D layout: VGPR r -> M = r + (lane>=16 ? 8 : 0), N = lane&15.
    const int n  = lane & 15;
    const int mb = (lane >> 4) << 3;
    const float bias = b_out[v0 + n];
    #pragma unroll
    for (int r = 0; r < 8; ++r)
        table[(size_t)(t0 + mb + r) * V_DIM + v0 + n] = c[r] + bias;
}

// ---------------------------------------------------------------------------
// Kernel 3: streaming gather. out[row,:] = table[x[row],:]. 4 rows per block,
// 64 lanes x float4 per row. Table (256 KiB) stays L2-resident; the 64 MiB
// output stream uses non-temporal stores. This is the only HBM-significant
// kernel: ~64 MiB / 23.3 TB/s ~ 2.9 us.
// ---------------------------------------------------------------------------
__global__ __launch_bounds__(256) void gather_kernel(
    const int* __restrict__ x, const float* __restrict__ table,
    float* __restrict__ out)
{
    const int rowidx = blockIdx.x * 4 + (threadIdx.x >> 6);
    const int lane   = threadIdx.x & 63;
    const int t      = x[rowidx] & 255;
    f4 val = ((const f4*)(table + (size_t)t * V_DIM))[lane];
    __builtin_nontemporal_store(val, (f4*)(out + (size_t)rowidx * V_DIM) + lane);
}

// ---------------------------------------------------------------------------
extern "C" void kernel_launch(void* const* d_in, const int* in_sizes, int n_in,
                              void* d_out, int out_size, void* d_ws, size_t ws_size,
                              hipStream_t stream)
{
    const int*   x      = (const int*)d_in[0];
    const float* embed  = (const float*)d_in[1];
    const float* freq   = (const float*)d_in[2];
    const float* ln_w   = (const float*)d_in[3];
    const float* ln_b   = (const float*)d_in[4];
    const float* W_out  = (const float*)d_in[5];
    const float* b_out  = (const float*)d_in[6];
    float*       out    = (float*)d_out;

    // Workspace layout (768 KiB total)
    char* ws = (char*)d_ws;
    _Float16* hn_hi = (_Float16*)(ws + 0);
    _Float16* hn_lo = (_Float16*)(ws + 131072);
    _Float16* w_hi  = (_Float16*)(ws + 262144);
    _Float16* w_lo  = (_Float16*)(ws + 393216);
    float*    table = (float*)   (ws + 524288);

    prep_kernel<<<512, 256, 0, stream>>>(embed, freq, ln_w, ln_b, W_out,
                                         hn_hi, hn_lo, w_hi, w_lo);
    table_gemm_kernel<<<32, 256, 0, stream>>>(hn_hi, hn_lo, w_hi, w_lo,
                                              b_out, table);
    gather_kernel<<<NROWS / 4, 256, 0, stream>>>(x, table, out);
}
